// TransformerBlock_38001870635315
// MI455X (gfx1250) — matrix operand order repaired and checked by reading, not compiled, not run on hardware
//
#include <hip/hip_runtime.h>
#include <hip/hip_bf16.h>

// ---------------------------------------------------------------------------
// Transformer block for MI455X (gfx1250), wave32 + WMMA f32_16x16x32_f16.
// Staging uses the CDNA5 Tensor Data Mover (tensor_load_to_lds + TENSORcnt)
// when available, falling back to global_load_async_to_lds_b128 (ASYNCcnt),
// then to plain copies. GEMM LDS tiles are double-buffered.
// ---------------------------------------------------------------------------

typedef __attribute__((ext_vector_type(16))) _Float16 v16h;
typedef __attribute__((ext_vector_type(8)))  _Float16 v8h;
typedef __attribute__((ext_vector_type(8)))  float    v8f;

constexpr int kB  = 2;
constexpr int kS  = 2048;
constexpr int kD  = 1024;
constexpr int kH  = 16;
constexpr int kDK = 64;
constexpr int kBS = kB * kS;   // 4096 token rows

// ----- CDNA5 data-movement feature detection (device pass only) ------------
#if defined(__AMDGCN__) && defined(__gfx1250__) && __has_builtin(__builtin_amdgcn_tensor_load_to_lds)
#define HAVE_TDM 1
#else
#define HAVE_TDM 0
#endif
#if defined(__AMDGCN__) && defined(__gfx1250__) && __has_builtin(__builtin_amdgcn_global_load_async_to_lds_b128)
#define HAVE_ASYNC 1
#else
#define HAVE_ASYNC 0
#endif

#if HAVE_TDM
typedef __attribute__((ext_vector_type(4))) unsigned int u32x4;
typedef __attribute__((ext_vector_type(8))) int          i32x8;
typedef __attribute__((ext_vector_type(4))) int          i32x4;

// Issue one TDM 2-D tile load: rows x rowlen(halfs) from row-major global
// (row stride gstride halfs) into LDS with +16B padding per row, so the LDS
// row stride is rowlen+8 halfs. rowlen must be 32 or 64.
__device__ inline void tdm_load_tile(const _Float16* g, int gstride,
                                     int rows, int rowlen, _Float16* lds) {
  const unsigned long long ga = (unsigned long long)(size_t)g;
  const unsigned lptr =
      (unsigned)(size_t)((__attribute__((address_space(3))) const void*)lds);
  u32x4 g0;
  g0[0] = 1u;                                   // count=1 (valid user D#)
  g0[1] = lptr;                                 // lds_addr
  g0[2] = (unsigned)(ga & 0xFFFFFFFFu);         // global_addr[31:0]
  g0[3] = (unsigned)((ga >> 32) & 0x01FFFFFFu)  // global_addr[56:32]
        | (2u << 30);                           // type = 2 ("image")
  const unsigned interval_code = (rowlen == 64) ? 4u : 3u;  // 32 / 16 DWORDs
  unsigned w0 = (1u << 16)                      // data_size = 2 bytes
              | (1u << 20)                      // pad_enable
              | (interval_code << 22)           // pad after each row
              | (3u << 25);                     // pad_amount = 4 DWORDs (16B)
  unsigned w1 = ((unsigned)rowlen) << 16;       // tensor_dim0[15:0] in [31:16]
  unsigned w2 = ((unsigned)rows) << 16;         // dim0 hi=0; tensor_dim1 lo16
  unsigned w3 = ((unsigned)rowlen) << 16;       // dim1 hi=0; tile_dim0
  unsigned w4 = (unsigned)rows;                 // tile_dim1; tile_dim2=0
  unsigned w5 = (unsigned)gstride;              // tensor_dim0_stride[31:0]
  i32x8 g1;
  g1[0] = (int)w0; g1[1] = (int)w1; g1[2] = (int)w2; g1[3] = (int)w3;
  g1[4] = (int)w4; g1[5] = (int)w5; g1[6] = 0;   g1[7] = 0;
  const i32x4 z4 = {0, 0, 0, 0};
#if __clang_major__ >= 23
  const i32x8 z8 = {0, 0, 0, 0, 0, 0, 0, 0};
  __builtin_amdgcn_tensor_load_to_lds(g0, g1, z4, z4, z8, 0);
#else
  __builtin_amdgcn_tensor_load_to_lds(g0, g1, z4, z4, 0);
#endif
}
#endif  // HAVE_TDM

// 16-byte global -> LDS copy (async when available).
// Builtin signature (from hipcc diagnostic): arg0 = int4 addrspace(1)*
// (global source, non-const), arg1 = int4 addrspace(3)* (LDS dest),
// then imm offset, imm cpol.
typedef __attribute__((__vector_size__(16))) int b128v;

__device__ inline void cp16(_Float16* l, const _Float16* g) {
#if HAVE_ASYNC
  __builtin_amdgcn_global_load_async_to_lds_b128(
      (__attribute__((address_space(1))) b128v*)g,
      (__attribute__((address_space(3))) b128v*)l, 0, 0);
#else
  *(v8h*)l = *(const v8h*)g;
#endif
}

// Wait for outstanding staging transfers issued by this wave.
__device__ inline void wait_stage() {
#if HAVE_TDM
  __builtin_amdgcn_s_wait_tensorcnt(0);
#elif HAVE_ASYNC
#if __has_builtin(__builtin_amdgcn_s_wait_asynccnt)
  __builtin_amdgcn_s_wait_asynccnt(0);
#else
  asm volatile("s_wait_asynccnt 0x0" ::: "memory");
#endif
#endif
}

__device__ inline v8f zero8() {
  v8f z;
#pragma unroll
  for (int i = 0; i < 8; ++i) z[i] = 0.0f;
  return z;
}

__device__ inline v16h join16(v8h a, v8h b) {
  return __builtin_shufflevector(a, b, 0,1,2,3,4,5,6,7,8,9,10,11,12,13,14,15);
}

// A fragment (16x32 f16, M x K): lane holds row M = lane&15.
// lanes 0-15: K chunks [0,8) and [16,24); lanes 16-31: [8,16) and [24,32).
__device__ inline v16h a_frag(const _Float16* t, int ld, int kbase, int lane) {
  const int lo = lane & 15, hi = lane >> 4;
  const _Float16* p = t + lo * ld + kbase + hi * 8;
  return join16(*(const v8h*)p, *(const v8h*)(p + 16));
}

// B fragment (32x16 f16, K x N): lane holds column N = lane&15.
// lanes 0-15: K chunks [0,8) and [16,24); lanes 16-31: [8,16) and [24,32).
__device__ inline v16h b_frag(const _Float16* t, int ld, int kbase, int lane) {
  const int lo = lane & 15, hi = lane >> 4;
  const _Float16* p = t + lo * ld + kbase + hi * 8;
  return join16(*(const v8h*)p, *(const v8h*)(p + 16));
}

__device__ inline v8f wmma_f16(v16h a, v16h b, v8f c) {
  return __builtin_amdgcn_wmma_f32_16x16x32_f16(false, a, false, b,
                                                (short)0, c, false, false);
}

// ---------------------------------------------------------------------------
// Kernel: f32 -> f16 weight convert (weights are N x K row-major, kept so)
// ---------------------------------------------------------------------------
__global__ __launch_bounds__(256)
void cvt_f16_kernel(const float* __restrict__ in, _Float16* __restrict__ out) {
  const int i = (blockIdx.x * 256 + threadIdx.x) * 4;
#pragma unroll
  for (int j = 0; j < 4; ++j) out[i + j] = (_Float16)in[i + j];
}

// ---------------------------------------------------------------------------
// Kernel: RMSNorm (+ optional RoPE) -> f16 row-major [rows, 1024]
// ---------------------------------------------------------------------------
template<bool ROPE>
__global__ __launch_bounds__(128)
void rmsnorm_kernel(const float* __restrict__ x, const float* __restrict__ g,
                    _Float16* __restrict__ out) {
  const int row = blockIdx.x;
  const int s = row & (kS - 1);
  const float* xr = x + (size_t)row * kD;

  float ss = 0.0f;
  for (int i = threadIdx.x; i < kD; i += 128) {
    const float v = xr[i];
    ss = fmaf(v, v, ss);
  }
#pragma unroll
  for (int m = 16; m >= 1; m >>= 1) ss += __shfl_xor(ss, m, 32);
  __shared__ float red[4];
  if ((threadIdx.x & 31) == 0) red[threadIdx.x >> 5] = ss;
  __syncthreads();
  const float tot = red[0] + red[1] + red[2] + red[3];
  const float rinv = 1.0f / sqrtf(tot * (1.0f / (float)kD) + 1.0e-6f);

  for (int i = threadIdx.x; i < kD; i += 128) {
    float o;
    const float xv = xr[i] * g[i] * rinv;
    if (ROPE) {
      const int d = i & 63;          // pos within head
      const int hb0 = i & ~63;       // head base
      const int j = d & 31;          // theta index
      const float theta = expf(-(float)j * (9.210340371976184f / 32.0f));
      const float fr = (float)s * theta;
      const float c = cosf(fr), sn = sinf(fr);
      if (d < 32) {
        const float pv = xr[hb0 + d + 32] * g[hb0 + d + 32] * rinv;
        o = xv * c - pv * sn;
      } else {
        const float pv = xr[hb0 + d - 32] * g[hb0 + d - 32] * rinv;
        o = xv * c + pv * sn;
      }
    } else {
      o = xv;
    }
    out[(size_t)row * kD + i] = (_Float16)o;
  }
}

// ---------------------------------------------------------------------------
// WMMA GEMM: C[M,N] = A[M,K](f16) * W[N,K](f16)^T + bias, templated epilogue.
// Block tile 128x128 (256 thr = 8 waves, wave tile 64x32 = 4x2 WMMA tiles).
// Double-buffered LDS staging via TDM / async-to-LDS.
// ---------------------------------------------------------------------------
enum { EPI_QKV = 0, EPI_WO = 1, EPI_RELU = 2, EPI_OUT = 3 };

template<int EPI>
__global__ __launch_bounds__(256)
void gemm_epi(const _Float16* __restrict__ A, const _Float16* __restrict__ Wt,
              const float* __restrict__ bias, const float* __restrict__ resid,
              void* __restrict__ outp, int M, int N, int K) {
  constexpr int LDT = 40;  // 32 halfs + 16B pad (matches TDM pad config)
  __shared__ __align__(16) _Float16 sA[2][128 * LDT];
  __shared__ __align__(16) _Float16 sB[2][128 * LDT];

  const int tid  = threadIdx.x;
  const int lane = tid & 31;
  const int wave = tid >> 5;
  const int wm = wave >> 2;   // 0..1
  const int wn = wave & 3;    // 0..3
  const int m0 = blockIdx.x * 128;
  const int n0 = blockIdx.y * 128;
  const int lo = lane & 15, hi = lane >> 4;

  v8f acc[4][2];
#pragma unroll
  for (int i = 0; i < 4; ++i) { acc[i][0] = zero8(); acc[i][1] = zero8(); }

  auto stage = [&](int buf, int k0) {
#if HAVE_TDM
    if (wave == 0) {
      tdm_load_tile(&A[(size_t)m0 * K + k0],  K, 128, 32, &sA[buf][0]);
      tdm_load_tile(&Wt[(size_t)n0 * K + k0], K, 128, 32, &sB[buf][0]);
    }
#else
    const int srow = tid >> 2;          // 0..63
    const int skc  = (tid & 3) * 8;     // 0,8,16,24
#pragma unroll
    for (int p = 0; p < 2; ++p) {
      const int row = srow + p * 64;
      cp16(&sA[buf][row * LDT + skc], &A[(size_t)(m0 + row) * K + k0 + skc]);
      cp16(&sB[buf][row * LDT + skc], &Wt[(size_t)(n0 + row) * K + k0 + skc]);
    }
#endif
  };

  stage(0, 0);
  const int nk = K / 32;
  for (int t = 0; t < nk; ++t) {
    wait_stage();
    __syncthreads();
    if (t + 1 < nk) stage((t + 1) & 1, (t + 1) * 32);

    const _Float16* tA = &sA[t & 1][0];
    const _Float16* tB = &sB[t & 1][0];
    v16h af[4], bf[2];
#pragma unroll
    for (int i = 0; i < 4; ++i)
      af[i] = a_frag(&tA[(wm * 64 + i * 16) * LDT], LDT, 0, lane);
#pragma unroll
    for (int j = 0; j < 2; ++j)
      bf[j] = b_frag(&tB[(wn * 32 + j * 16) * LDT], LDT, 0, lane);
#pragma unroll
    for (int i = 0; i < 4; ++i)
#pragma unroll
      for (int j = 0; j < 2; ++j)
        acc[i][j] = wmma_f16(af[i], bf[j], acc[i][j]);
    __syncthreads();
  }

  // Epilogue. C layout: element (M = r + hi*8, N = lo) in acc[.][.][r].
#pragma unroll
  for (int i = 0; i < 4; ++i) {
#pragma unroll
    for (int j = 0; j < 2; ++j) {
      const int col = n0 + wn * 32 + j * 16 + lo;
      const float bv = bias[col];
#pragma unroll
      for (int r = 0; r < 8; ++r) {
        const int row = m0 + wm * 64 + i * 16 + r + hi * 8;
        const float v = acc[i][j][r] + bv;
        if constexpr (EPI == EPI_QKV) {
          const int bb = row >> 11, sidx = row & (kS - 1);
          const int hh = col >> 6,  dd   = col & 63;
          ((_Float16*)outp)[(((size_t)bb * kH + hh) * kS + sidx) * kDK + dd] =
              (_Float16)v;
        } else if constexpr (EPI == EPI_WO) {
          ((float*)outp)[(size_t)row * N + col] = 2.0f * v;  // x = x + x
        } else if constexpr (EPI == EPI_RELU) {
          ((_Float16*)outp)[(size_t)row * N + col] = (_Float16)fmaxf(v, 0.0f);
        } else {  // EPI_OUT
          ((float*)outp)[(size_t)row * N + col] =
              v + resid[(size_t)row * N + col];
        }
      }
    }
  }
}

// ---------------------------------------------------------------------------
// Flash attention: block = 4 waves, each wave owns 16 query rows (block: 64).
// K tile staged via TDM/async; V tile transposed into LDS; online softmax.
// ---------------------------------------------------------------------------
__global__ __launch_bounds__(128)
void attn_kernel(const _Float16* __restrict__ Qb, const _Float16* __restrict__ Kb,
                 const _Float16* __restrict__ Vb, _Float16* __restrict__ ctx) {
  constexpr int LDK = 72;  // 64 + pad, 144B row stride (matches TDM pad)
  __shared__ __align__(16) _Float16 sK[64 * LDK];
  __shared__ __align__(16) _Float16 sVT[64 * LDK];
  __shared__ __align__(16) _Float16 sP[4][16 * 64];

  const int tid  = threadIdx.x;
  const int wave = tid >> 5, lane = tid & 31;
  const int lo = lane & 15, hi = lane >> 4;
  const int qt = blockIdx.x, h = blockIdx.y, b = blockIdx.z;
  const size_t hbase = ((size_t)b * kH + h) * (size_t)kS * kDK;
  const int q0 = qt * 64 + wave * 16;

  const v16h qf0 = a_frag(Qb + hbase + (size_t)q0 * kDK, kDK, 0,  lane);
  const v16h qf1 = a_frag(Qb + hbase + (size_t)q0 * kDK, kDK, 32, lane);

  v8f cacc[4];
  float mrow[8], lrow[8];
#pragma unroll
  for (int i = 0; i < 4; ++i) cacc[i] = zero8();
#pragma unroll
  for (int r = 0; r < 8; ++r) { mrow[r] = -3.0e38f; lrow[r] = 0.0f; }

  const int ntiles = qt + 1;  // causal: key tiles 0..qt
  for (int kt = 0; kt < ntiles; ++kt) {
    __syncthreads();
    // ---- stage K tile (TDM / async) ----
#if HAVE_TDM
    if (wave == 0)
      tdm_load_tile(Kb + hbase + (size_t)kt * 64 * kDK, kDK, 64, 64, sK);
#else
    {
      const int row = tid >> 1, c0 = (tid & 1) * 32;
      const _Float16* kg = Kb + hbase + ((size_t)kt * 64 + row) * kDK + c0;
#pragma unroll
      for (int j = 0; j < 4; ++j)
        cp16(&sK[row * LDK + c0 + j * 8], kg + j * 8);
    }
#endif
    // ---- stage V tile transposed (register transpose + LDS scatter) ----
    {
      const int row = tid >> 1, c0 = (tid & 1) * 32;
      const _Float16* vg = Vb + hbase + ((size_t)kt * 64 + row) * kDK + c0;
      v8h vv[4];
#pragma unroll
      for (int j = 0; j < 4; ++j) vv[j] = *(const v8h*)(vg + j * 8);
#pragma unroll
      for (int j = 0; j < 4; ++j)
#pragma unroll
        for (int e = 0; e < 8; ++e)
          sVT[(c0 + j * 8 + e) * LDK + row] = vv[j][e];
    }
    wait_stage();
    __syncthreads();

    // ---- scores: 4 subtiles of 16 keys; 2 WMMAs each (K-dim = 64) ----
    v8f sc[4];
#pragma unroll
    for (int st = 0; st < 4; ++st) {
      v8f a = zero8();
      a = wmma_f16(qf0, b_frag(&sK[st * 16 * LDK], LDK, 0,  lane), a);
      a = wmma_f16(qf1, b_frag(&sK[st * 16 * LDK], LDK, 32, lane), a);
      sc[st] = a;
    }
    // ---- scale + causal mask ----
#pragma unroll
    for (int st = 0; st < 4; ++st) {
      const int key = kt * 64 + st * 16 + lo;
#pragma unroll
      for (int r = 0; r < 8; ++r) {
        float v = sc[st][r] * 0.125f;  // 1/sqrt(64)
        if (key > q0 + r + hi * 8) v = -1.0e9f;
        sc[st][r] = v;
      }
    }
    // ---- online softmax; write P (f16) to per-wave LDS tile ----
    _Float16* sPw = &sP[wave][0];
#pragma unroll
    for (int r = 0; r < 8; ++r) {
      float m = sc[0][r];
#pragma unroll
      for (int st = 1; st < 4; ++st) m = fmaxf(m, sc[st][r]);
#pragma unroll
      for (int d = 8; d >= 1; d >>= 1) m = fmaxf(m, __shfl_xor(m, d, 32));
      const float mn = fmaxf(mrow[r], m);
      const float corr = __expf(mrow[r] - mn);
      mrow[r] = mn;
      float rs = 0.0f;
#pragma unroll
      for (int st = 0; st < 4; ++st) {
        const float p = __expf(sc[st][r] - mn);
        rs += p;
        sPw[(r + hi * 8) * 64 + st * 16 + lo] = (_Float16)p;
      }
#pragma unroll
      for (int d = 8; d >= 1; d >>= 1) rs += __shfl_xor(rs, d, 32);
      lrow[r] = lrow[r] * corr + rs;
#pragma unroll
      for (int dt = 0; dt < 4; ++dt) cacc[dt][r] *= corr;
    }
    // ---- ctx += P * V ----
    const v16h pa0 = a_frag(sPw, 64, 0,  lane);
    const v16h pa1 = a_frag(sPw, 64, 32, lane);
#pragma unroll
    for (int dt = 0; dt < 4; ++dt) {
      cacc[dt] = wmma_f16(pa0, b_frag(&sVT[dt * 16 * LDK], LDK, 0,  lane), cacc[dt]);
      cacc[dt] = wmma_f16(pa1, b_frag(&sVT[dt * 16 * LDK], LDK, 32, lane), cacc[dt]);
    }
  }

  // normalize + store ctx as f16 row-major [B*S, D], col = h*64 + d
#pragma unroll
  for (int dt = 0; dt < 4; ++dt) {
#pragma unroll
    for (int r = 0; r < 8; ++r) {
      const int qrow = q0 + r + hi * 8;
      const int col = h * kDK + dt * 16 + lo;
      ctx[((size_t)b * kS + qrow) * kD + col] = (_Float16)(cacc[dt][r] / lrow[r]);
    }
  }
}

// ---------------------------------------------------------------------------
// Host-side launcher
// ---------------------------------------------------------------------------
extern "C" void kernel_launch(void* const* d_in, const int* in_sizes, int n_in,
                              void* d_out, int out_size, void* d_ws, size_t ws_size,
                              hipStream_t stream) {
  (void)in_sizes; (void)n_in; (void)out_size; (void)ws_size;
  const float* x  = (const float*)d_in[0];
  // d_in[1] = mask (causal, applied analytically)
  const float* Wq = (const float*)d_in[2];
  const float* bq = (const float*)d_in[3];
  const float* Wk = (const float*)d_in[4];
  const float* bk = (const float*)d_in[5];
  const float* Wv = (const float*)d_in[6];
  const float* bv = (const float*)d_in[7];
  const float* Wo = (const float*)d_in[8];
  const float* bo = (const float*)d_in[9];
  const float* W1 = (const float*)d_in[10];
  const float* b1 = (const float*)d_in[11];
  const float* W2 = (const float*)d_in[12];
  const float* b2 = (const float*)d_in[13];
  const float* g  = (const float*)d_in[14];

  char* w = (char*)d_ws;
  const size_t MB = 1024ull * 1024ull;
  _Float16* wqh  = (_Float16*)(w +  0 * MB);
  _Float16* wkh  = (_Float16*)(w +  2 * MB);
  _Float16* wvh  = (_Float16*)(w +  4 * MB);
  _Float16* woh  = (_Float16*)(w +  6 * MB);
  _Float16* w1h  = (_Float16*)(w +  8 * MB);
  _Float16* w2h  = (_Float16*)(w + 10 * MB);
  _Float16* xn   = (_Float16*)(w + 12 * MB);
  _Float16* qb   = (_Float16*)(w + 20 * MB);
  _Float16* kbuf = (_Float16*)(w + 28 * MB);
  _Float16* vbuf = (_Float16*)(w + 36 * MB);
  _Float16* ctxb = (_Float16*)(w + 44 * MB);
  float*    x2   = (float*)   (w + 52 * MB);
  _Float16* hbuf = (_Float16*)(w + 68 * MB);
  _Float16* a1   = (_Float16*)(w + 76 * MB);

  const int NW = kD * kD;
  dim3 cvb(256), cvg(NW / 1024);
  cvt_f16_kernel<<<cvg, cvb, 0, stream>>>(Wq, wqh);
  cvt_f16_kernel<<<cvg, cvb, 0, stream>>>(Wk, wkh);
  cvt_f16_kernel<<<cvg, cvb, 0, stream>>>(Wv, wvh);
  cvt_f16_kernel<<<cvg, cvb, 0, stream>>>(Wo, woh);
  cvt_f16_kernel<<<cvg, cvb, 0, stream>>>(W1, w1h);
  cvt_f16_kernel<<<cvg, cvb, 0, stream>>>(W2, w2h);

  rmsnorm_kernel<true><<<kBS, 128, 0, stream>>>(x, g, xn);

  dim3 gg(kBS / 128, kD / 128), gb(256);
  gemm_epi<EPI_QKV><<<gg, gb, 0, stream>>>(xn, wqh, bq, nullptr, qb,   kBS, kD, kD);
  gemm_epi<EPI_QKV><<<gg, gb, 0, stream>>>(xn, wkh, bk, nullptr, kbuf, kBS, kD, kD);
  gemm_epi<EPI_QKV><<<gg, gb, 0, stream>>>(xn, wvh, bv, nullptr, vbuf, kBS, kD, kD);

  attn_kernel<<<dim3(kS / 64, kH, kB), 128, 0, stream>>>(qb, kbuf, vbuf, ctxb);

  gemm_epi<EPI_WO><<<gg, gb, 0, stream>>>(ctxb, woh, bo, nullptr, x2, kBS, kD, kD);

  rmsnorm_kernel<false><<<kBS, 128, 0, stream>>>(x2, g, hbuf);

  gemm_epi<EPI_RELU><<<gg, gb, 0, stream>>>(hbuf, w1h, b1, nullptr, a1, kBS, kD, kD);
  gemm_epi<EPI_OUT><<<gg, gb, 0, stream>>>(a1, w2h, b2, x2, d_out, kBS, kD, kD);
}
